// HyperFNS_88304527606525
// MI455X (gfx1250) — matrix operand order repaired and checked by reading, not compiled, not run on hardware
//
#include <hip/hip_runtime.h>
#include <cstddef>

#define NN 127
#define NP 256
#define NPNP (NP * NP)
#define NB 16

typedef __attribute__((ext_vector_type(16))) _Float16 v16h;
typedef __attribute__((ext_vector_type(8)))  float    v8f;
typedef __attribute__((ext_vector_type(4)))  unsigned int v4u;
typedef __attribute__((ext_vector_type(4)))  int v4i;
typedef __attribute__((ext_vector_type(8)))  int v8i;

#if __has_builtin(__builtin_amdgcn_tensor_load_to_lds) && \
    __has_builtin(__builtin_amdgcn_s_wait_tensorcnt)
#define HAVE_TDM 1
#else
#define HAVE_TDM 0
#endif

// ---------------- workspace layout (bytes) ----------------
static const size_t WS_X    = 0;                 // 16*127*127 f32 = 1,032,256 B
static const size_t WS_R    = (size_t)1  << 20;  // residual, same size
static const size_t WS_PART = (size_t)2  << 20;  // 1024 f32 partial sums
static const size_t WS_A    = (size_t)4  << 20;  // [16][4][256][256] float2 = 32 MB
static const size_t WS_B    = (size_t)40 << 20;  // [16][4][256][256] float2 = 32 MB

__device__ __forceinline__ int outer_iters(const int* ep) {
    int e = ep[0];
    int k = (e - 1) / 40 + 1;
    return k < 10 ? k : 10;
}

// ---------------- zero x ----------------
__global__ void zero_kernel(float* p, int n) {
    int i = blockIdx.x * blockDim.x + threadIdx.x;
    if (i < n) p[i] = 0.0f;
}

// ---------------- 20 Jacobi sweeps, whole grid in LDS ----------------
__global__ __launch_bounds__(1024) void jacobi20_kernel(
    float* __restrict__ x, const float* __restrict__ f,
    const float* __restrict__ kA, const int* ep, int it)
{
    if (it >= outer_iters(ep)) return;
    int b = blockIdx.x;
    __shared__ float xs[NN * NN];
    float k[9];
#pragma unroll
    for (int i = 0; i < 9; ++i) k[i] = kA[b * 9 + i];
    float tau = 0.5f / k[4];
    const float* fb = f + b * NN * NN;
    float* xb = x + b * NN * NN;
    int t = threadIdx.x;
    for (int p = t; p < NN * NN; p += 1024) xs[p] = xb[p];
    __syncthreads();
    for (int step = 0; step < 20; ++step) {
        float nv[16];
        int c = 0;
        for (int p = t; p < NN * NN; p += 1024, ++c) {
            int i = p / NN, j = p - i * NN;
            float s = 0.0f;
#pragma unroll
            for (int u = 0; u < 3; ++u)
#pragma unroll
                for (int v = 0; v < 3; ++v) {
                    int ii = i + u - 1, jj = j + v - 1;
                    float xv = (ii >= 0 && ii < NN && jj >= 0 && jj < NN)
                                   ? xs[ii * NN + jj] : 0.0f;
                    s += k[u * 3 + v] * xv;
                }
            nv[c] = xs[p] + tau * (fb[p] - s);
        }
        __syncthreads();
        c = 0;
        for (int p = t; p < NN * NN; p += 1024, ++c) xs[p] = nv[c];
        __syncthreads();
    }
    for (int p = t; p < NN * NN; p += 1024) xb[p] = xs[p];
}

// ---------------- r = f - A x ----------------
__global__ void residual_kernel(float* __restrict__ r, const float* __restrict__ x,
                                const float* __restrict__ f, const float* __restrict__ kA,
                                const int* ep, int it)
{
    if (it >= outer_iters(ep)) return;
    int p = blockIdx.x * blockDim.x + threadIdx.x;
    if (p >= NB * NN * NN) return;
    int b = p / (NN * NN);
    int q = p - b * NN * NN;
    int i = q / NN, j = q - i * NN;
    const float* xb = x + b * NN * NN;
    const float* kb = kA + b * 9;
    float s = 0.0f;
#pragma unroll
    for (int u = 0; u < 3; ++u)
#pragma unroll
        for (int v = 0; v < 3; ++v) {
            int ii = i + u - 1, jj = j + v - 1;
            float xv = (ii >= 0 && ii < NN && jj >= 0 && jj < NN)
                           ? xb[ii * NN + jj] : 0.0f;
            s += kb[u * 3 + v] * xv;
        }
    r[p] = f[p] - s;
}

// ------- odd-symmetric expand + fftshift sign fold: A[c0] = expand(r)*(-1)^(y+x) -------
__global__ void expand_kernel(float2* __restrict__ A, const float* __restrict__ r,
                              const int* ep, int it)
{
    if (it >= outer_iters(ep)) return;
    int idx = blockIdx.x * blockDim.x + threadIdx.x;
    if (idx >= NB * NPNP) return;
    int b = idx >> 16;
    int p = idx & (NPNP - 1);
    int y = p >> 8;
    int xx = p & 255;
    float v = 0.0f;
    if (y != 0 && y != 128 && xx != 0 && xx != 128) {
        int i, j;
        float sg = 1.0f;
        if (y < 128)  i = y - 1;  else { i = 255 - y;  sg = -sg; }
        if (xx < 128) j = xx - 1; else { j = 255 - xx; sg = -sg; }
        v = sg * r[(b * NN + i) * NN + j];
    }
    if ((y + xx) & 1) v = -v;  // fold fftshift(ifft2(.)) into input signs
    A[(size_t)b * 4 * NPNP + p] = make_float2(v, 0.0f);
}

// ---------------- TDM: 2D strided tile -> LDS (D# per ISA ch.8) ----------------
#if HAVE_TDM
__device__ __forceinline__ void tdm_load_2d(unsigned lds_off, const void* gaddr,
                                            unsigned tile0, unsigned tile1,
                                            unsigned dim0, unsigned dim1,
                                            unsigned long long stride0)
{
    unsigned long long ga = (unsigned long long)gaddr;
    v4u g0;
    g0.x = 1u;                                        // count=1 (valid), user mode
    g0.y = lds_off;                                   // lds_addr (bytes)
    g0.z = (unsigned)(ga & 0xffffffffu);              // global_addr[31:0]
    g0.w = (unsigned)((ga >> 32) & 0x01ffffffu) | (2u << 30); // ga[56:32] | type=2
    v8i g1;
    g1[0] = (int)(3u << 16);                          // data_size=3 (8B), mask=0
    g1[1] = (int)((dim0 & 0xffffu) << 16);            // tensor_dim0[15:0]
    g1[2] = (int)((dim0 >> 16) | ((dim1 & 0xffffu) << 16));
    g1[3] = (int)((dim1 >> 16) | ((tile0 & 0xffffu) << 16));
    g1[4] = (int)(tile1 & 0xffffu);                   // tile_dim1 (tile_dim2=0)
    g1[5] = (int)(unsigned)(stride0 & 0xffffffffull); // tensor_dim0_stride[31:0]
    g1[6] = (int)(unsigned)((stride0 >> 32) & 0xffffull);
    g1[7] = 0;                                        // tensor_dim1_stride = 0
    v4i g2 = {0, 0, 0, 0};
    v4i g3 = {0, 0, 0, 0};
#if __clang_major__ >= 23
    v8i g1b = {0, 0, 0, 0, 0, 0, 0, 0};
    __builtin_amdgcn_tensor_load_to_lds(g0, g1, g2, g3, g1b, 0);
#else
    __builtin_amdgcn_tensor_load_to_lds(g0, g1, g2, g3, 0);
#endif
}
#endif

// ---------------- 256-pt radix-2 FFT in LDS, one line per block ----------------
// dir = +1 -> inverse kernel sign (with scale 1/256 per pass), dir = -1 -> forward
__global__ void fft256_kernel(float2* __restrict__ buf, float dir, float scale,
                              int axis, const int* ep, int it)
{
    if (it >= outer_iters(ep)) return;
    int l = blockIdx.x;
    int b = l >> 8;
    int q = l & 255;
    float2* base = buf + (size_t)b * 4 * NPNP;  // channel 0
    int stride = axis ? NP : 1;
    int off = axis ? q : q * NP;
    __shared__ float2 Sg[NP];   // staging (natural order)
    __shared__ float2 S[NP];    // working (bit-reversed -> natural)
    int t = threadIdx.x;
#if HAVE_TDM
    // One wave issues the DMA: rows are contiguous (tile 256x1), columns are a
    // strided gather (tile 1x256, row stride = 256 elements of 8B).
    if (t < 32) {
        unsigned lds_off = (unsigned)(unsigned long long)(void*)Sg;
        const float2* gsrc = base + off;
        if (axis == 0) tdm_load_2d(lds_off, gsrc, 256u, 1u, 256u, 1u, 256ull);
        else           tdm_load_2d(lds_off, gsrc, 1u, 256u, 1u, 256u, 256ull);
        __builtin_amdgcn_s_wait_tensorcnt(0);
    }
    __syncthreads();
#else
    for (int i = t; i < NP; i += 128) Sg[i] = base[off + i * stride];
    __syncthreads();
#endif
    for (int i = t; i < NP; i += 128) {
        int rb = __brev((unsigned)i) >> 24;
        S[rb] = Sg[i];
    }
    __syncthreads();
    for (int m = 2; m <= NP; m <<= 1) {
        int half = m >> 1;
        int j = t & (half - 1);
        int g = t / half;
        int i0 = g * m + j;
        float ang = dir * 6.28318530717958647692f * (float)j / (float)m;
        float sn, cs;
        __sincosf(ang, &sn, &cs);
        float2 u = S[i0];
        float2 v = S[i0 + half];
        float2 wv = make_float2(v.x * cs - v.y * sn, v.x * sn + v.y * cs);
        S[i0]        = make_float2(u.x + wv.x, u.y + wv.y);
        S[i0 + half] = make_float2(u.x - wv.x, u.y - wv.y);
        __syncthreads();
    }
    for (int i = t; i < NP; i += 128) {
        float2 v = S[i];
        base[off + i * stride] = make_float2(v.x * scale, v.y * scale);
    }
}

// ---------------- complex grouped 3x3 conv as WMMA implicit GEMM ----------------
// One wave computes a 16-pixel row segment for all cout (complex) channels.
// A: 16 pixels x K (K = 2*CI*9, re/im interleaved im2col), B: K x (2*CO),
// f16 operands, f32 accumulation via v_wmma_f32_16x16x32_f16.
// CI/CO/ADJ are compile-time so the K-chunk loop fully unrolls.
template <int CI, int CO, int ADJ>
__global__ __launch_bounds__(128) void conv_wmma_kernel(
    const float2* __restrict__ in, float2* __restrict__ out,
    const float* __restrict__ wr, const float* __restrict__ wi,
    const int* ep, int it)
{
    if (it >= outer_iters(ep)) return;
    constexpr int Ktot = CI * 18;
    int wave = blockIdx.x * (blockDim.x >> 5) + (threadIdx.x >> 5);
    int lane = threadIdx.x & 31;
    int b  = wave >> 12;
    int rem = wave & 4095;
    int y  = rem >> 4;
    int x0 = (rem & 15) << 4;
    int hi = lane >> 4;
    int ln = lane & 15;
    const float2* inb = in + (size_t)b * 4 * NPNP;
    if (y + 1 < NP) __builtin_prefetch(&inb[(y + 1) * NP + x0], 0, 1);
    int n = ln;            // output column of D
    int co_n = n >> 1;     // complex channel
    int q_n = n & 1;       // 0 = real, 1 = imag
    v8f acc = {};
#pragma unroll
    for (int kb = 0; kb < Ktot; kb += 32) {
        v16h Af = {};
        v16h Bf = {};
#pragma unroll
        for (int e = 0; e < 16; ++e) {
            // ---- A fragment element (16-bit A 16x32 wave32 layout) ----
            int ka = kb + ((e < 8) ? (hi * 8 + e) : (8 + hi * 8 + e));
            float av = 0.0f;
            if (ka < Ktot) {
                int p  = ka & 1;
                int r3 = ka >> 1;
                int ci = r3 / 9;
                int s9 = r3 - ci * 9;
                int uy = s9 / 3;
                int ux = s9 - uy * 3;
                int yy = y + uy - 1;
                int xx = x0 + ln + ux - 1;
                if (yy >= 0 && yy < NP && xx >= 0 && xx < NP) {
                    float2 tv = inb[(size_t)ci * NPNP + yy * NP + xx];
                    av = p ? tv.y : tv.x;
                }
            }
            Af[e] = (_Float16)av;
            // ---- B fragment element (16-bit B 32x16 wave32 layout) ----
            int kbb = kb + ((e < 8) ? (hi * 8 + e) : (8 + hi * 8 + e));
            float bv = 0.0f;
            if (kbb < Ktot && co_n < CO) {
                int p  = kbb & 1;
                int r3 = kbb >> 1;
                int ci = r3 / 9;
                int s9 = r3 - ci * 9;
                int uy = s9 / 3;
                int ux = s9 - uy * 3;
                float Wr, Wi;
                if (!ADJ) {
                    int wix = ((b * CO + co_n) * CI + ci) * 9 + uy * 3 + ux;
                    Wr = wr[wix]; Wi = wi[wix];
                } else {
                    // tconj(w): swap channel axes, transpose 3x3, conjugate
                    int wix = ((b * CI + ci) * CO + co_n) * 9 + ux * 3 + uy;
                    Wr = wr[wix]; Wi = -wi[wix];
                }
                // y_re = xr*Wr - xi*Wi ; y_im = xr*Wi + xi*Wr
                bv = (q_n == 0) ? ((p == 0) ? Wr : -Wi)
                                : ((p == 0) ? Wi :  Wr);
            }
            Bf[e] = (_Float16)bv;
        }
        acc = __builtin_amdgcn_wmma_f32_16x16x32_f16(
            false, Af, false, Bf, (short)0, acc, false, false);
    }
    // D layout: lane (hi,ln) elem v -> row M = hi*8+v (pixel), col N = ln
    float* outf = (float*)(out + (size_t)b * 4 * NPNP);
#pragma unroll
    for (int v = 0; v < 8; ++v) {
        int M = hi * 8 + v;
        if (co_n < CO)
            outf[((size_t)co_n * NPNP + (size_t)y * NP + (x0 + M)) * 2 + q_n] = acc[v];
    }
}

// ---------------- pointwise complex multiply by theta ----------------
__global__ void theta_mul_kernel(float2* __restrict__ buf, const float* __restrict__ tr,
                                 const float* __restrict__ ti, const int* ep, int it)
{
    if (it >= outer_iters(ep)) return;
    int idx = blockIdx.x * blockDim.x + threadIdx.x;
    if (idx >= NB * NPNP) return;
    int b = idx >> 16;
    int p = idx & (NPNP - 1);
    size_t o = (size_t)b * 4 * NPNP + p;
    float2 v = buf[o];
    float c = tr[idx], d = ti[idx];
    buf[o] = make_float2(v.x * c - v.y * d, v.x * d + v.y * c);
}

// ------- x += (-1)^(i+j) * Re(F)[i,j]  (ifftshift folded into output signs) -------
__global__ void accum_kernel(float* __restrict__ x, const float2* __restrict__ A,
                             const int* ep, int it)
{
    if (it >= outer_iters(ep)) return;
    int p = blockIdx.x * blockDim.x + threadIdx.x;
    if (p >= NB * NN * NN) return;
    int b = p / (NN * NN);
    int q = p - b * NN * NN;
    int i = q / NN, j = q - i * NN;
    float v = A[(size_t)b * 4 * NPNP + i * NP + j].x;
    if ((i + j) & 1) v = -v;
    x[p] += v;
}

// ---------------- deterministic norm reduction ----------------
__global__ void norm_part_kernel(float* __restrict__ parts, const float* __restrict__ x,
                                 const float* __restrict__ f, const float* __restrict__ kA)
{
    __shared__ float sr[256];
    __shared__ float sf[256];
    float ar = 0.0f, af = 0.0f;
    int tot = NB * NN * NN;
    for (int p = blockIdx.x * blockDim.x + threadIdx.x; p < tot;
         p += gridDim.x * blockDim.x) {
        int b = p / (NN * NN);
        int q = p - b * NN * NN;
        int i = q / NN, j = q - i * NN;
        const float* xb = x + b * NN * NN;
        const float* kb = kA + b * 9;
        float s = 0.0f;
#pragma unroll
        for (int u = 0; u < 3; ++u)
#pragma unroll
            for (int v = 0; v < 3; ++v) {
                int ii = i + u - 1, jj = j + v - 1;
                float xv = (ii >= 0 && ii < NN && jj >= 0 && jj < NN)
                               ? xb[ii * NN + jj] : 0.0f;
                s += kb[u * 3 + v] * xv;
            }
        float rv = f[p] - s;
        ar += rv * rv;
        af += f[p] * f[p];
    }
    int t = threadIdx.x;
    sr[t] = ar; sf[t] = af;
    __syncthreads();
    for (int s = 128; s > 0; s >>= 1) {
        if (t < s) { sr[t] += sr[t + s]; sf[t] += sf[t + s]; }
        __syncthreads();
    }
    if (t == 0) { parts[blockIdx.x] = sr[0]; parts[512 + blockIdx.x] = sf[0]; }
}

__global__ void finalize_kernel(float* __restrict__ out, const float* __restrict__ parts)
{
    __shared__ float sr[512];
    __shared__ float sf[512];
    int t = threadIdx.x;
    sr[t] = parts[t];
    sf[t] = parts[512 + t];
    __syncthreads();
    for (int s = 256; s > 0; s >>= 1) {
        if (t < s) { sr[t] += sr[t + s]; sf[t] += sf[t + s]; }
        __syncthreads();
    }
    if (t == 0) out[0] = sqrtf(sr[0] / sf[0]);
}

// ---------------- driver ----------------
extern "C" void kernel_launch(void* const* d_in, const int* in_sizes, int n_in,
                              void* d_out, int out_size, void* d_ws, size_t ws_size,
                              hipStream_t stream)
{
    const float* f   = (const float*)d_in[0];
    const float* kA  = (const float*)d_in[1];
    const float* w1r = (const float*)d_in[2];
    const float* w1i = (const float*)d_in[3];
    const float* w2r = (const float*)d_in[4];
    const float* w2i = (const float*)d_in[5];
    const float* w3r = (const float*)d_in[6];
    const float* w3i = (const float*)d_in[7];
    const float* thr = (const float*)d_in[8];
    const float* thi = (const float*)d_in[9];
    const int*   ep  = (const int*)d_in[10];
    float* out = (float*)d_out;

    char* ws = (char*)d_ws;
    float*  x     = (float*)(ws + WS_X);
    float*  r     = (float*)(ws + WS_R);
    float*  parts = (float*)(ws + WS_PART);
    float2* A     = (float2*)(ws + WS_A);
    float2* B     = (float2*)(ws + WS_B);

    const int NX = NB * NN * NN;                  // 258,064
    const int gridNX = (NX + 255) / 256;
    const int gridNP = (NB * NPNP + 255) / 256;   // 4096
    const float inv256 = 1.0f / 256.0f;

    zero_kernel<<<gridNX, 256, 0, stream>>>(x, NX);

    for (int it = 0; it < 10; ++it) {
        jacobi20_kernel<<<NB, 1024, 0, stream>>>(x, f, kA, ep, it);
        residual_kernel<<<gridNX, 256, 0, stream>>>(r, x, f, kA, ep, it);
        expand_kernel<<<gridNP, 256, 0, stream>>>(A, r, ep, it);
        // ifft2 (scale 1/256 per pass), fftshift already folded by expand
        fft256_kernel<<<NB * NP, 128, 0, stream>>>(A, +1.0f, inv256, 0, ep, it);
        fft256_kernel<<<NB * NP, 128, 0, stream>>>(A, +1.0f, inv256, 1, ep, it);
        // conv stack (forward)
        conv_wmma_kernel<1, 4, 0><<<16384, 128, 0, stream>>>(A, B, w1r, w1i, ep, it);
        conv_wmma_kernel<4, 4, 0><<<16384, 128, 0, stream>>>(B, A, w2r, w2i, ep, it);
        conv_wmma_kernel<4, 1, 0><<<16384, 128, 0, stream>>>(A, B, w3r, w3i, ep, it);
        theta_mul_kernel<<<gridNP, 256, 0, stream>>>(B, thr, thi, ep, it);
        // adjoint conv stack (tconj weights)
        conv_wmma_kernel<1, 4, 1><<<16384, 128, 0, stream>>>(B, A, w3r, w3i, ep, it);
        conv_wmma_kernel<4, 4, 1><<<16384, 128, 0, stream>>>(A, B, w2r, w2i, ep, it);
        conv_wmma_kernel<4, 1, 1><<<16384, 128, 0, stream>>>(B, A, w1r, w1i, ep, it);
        // fft2 (no scale), ifftshift folded into accum signs
        fft256_kernel<<<NB * NP, 128, 0, stream>>>(A, -1.0f, 1.0f, 0, ep, it);
        fft256_kernel<<<NB * NP, 128, 0, stream>>>(A, -1.0f, 1.0f, 1, ep, it);
        accum_kernel<<<gridNX, 256, 0, stream>>>(x, A, ep, it);
    }

    norm_part_kernel<<<512, 256, 0, stream>>>(parts, x, f, kA);
    finalize_kernel<<<1, 512, 0, stream>>>(out, parts);
}
